// SpatialAttention_84576495993002
// MI455X (gfx1250) — matrix operand order repaired and checked by reading, not compiled, not run on hardware
//
#include <hip/hip_runtime.h>
#include <hip/hip_bf16.h>
#include <math.h>

// ---------------------------------------------------------------------------
// SpatialAttention forward for MI455X (gfx1250), wave32 + WMMA bf16.
//
//   B=64, C=273, T=3000, CHOUT=270, NF=32, POS_DIM=2048
//   scores GEMM: M=17472, N=270(pad 272), K=2048           (19.3 GFLOP)
//   out   GEMM: 64 x [M=270(pad272), N=3000, K=273(pad288)] (28.3 GFLOP)
//
// HBM floor ~420MB => ~18us @ 23.3TB/s; bf16 WMMA keeps compute well below
// that, so the whole pipeline is memory-bound. bf16 quantization error
// averages down by sqrt(K) in each dot product (~0.1% on the output).
// ---------------------------------------------------------------------------

typedef __attribute__((ext_vector_type(16))) __bf16 bf16x16;
typedef __attribute__((ext_vector_type(8)))  __bf16 bf16x8;
typedef __attribute__((ext_vector_type(8)))  float  f32x8;

#define NB   64
#define NC   273
#define NT   3000
#define NO   270
#define NOP  272      // N padded to 17 tiles of 16
#define NCP  288      // K padded to 9 steps of 32
#define KPOS 2048
#define NROWS (NB*NC) // 17472 = 1092 * 16

static __device__ __forceinline__ bf16x16 cat8(bf16x8 lo, bf16x8 hi) {
    return __builtin_shufflevector(lo, hi, 0,1,2,3,4,5,6,7,8,9,10,11,12,13,14,15);
}

static __device__ __forceinline__ f32x8 wmma_bf16(bf16x16 a, bf16x16 b, f32x8 c) {
    return __builtin_amdgcn_wmma_f32_16x16x32_bf16(
        /*neg_a=*/false, a, /*neg_b=*/false, b,
        /*c_mod=*/(short)0, c, /*reuse_a=*/false, /*reuse_b=*/false);
}

// ---------------------------------------------------------------------------
// Kernel 0: heads f32 -> bf16 (272 rows, rows >=270 zeroed); zero-fill the
// K-padding columns c in [273,288) of the transposed-weights workspace.
// ---------------------------------------------------------------------------
__global__ __launch_bounds__(256) void sa_prep_kernel(
    const float* __restrict__ heads, __bf16* __restrict__ headsBF,
    __bf16* __restrict__ wsW) {
    int gid = blockIdx.x * 256 + threadIdx.x;
    if (gid < NOP * KPOS) {
        int o = gid >> 11;          // row (2048 per row)
        int k = gid & (KPOS - 1);
        float v = (o < NO) ? heads[o * KPOS + k] : 0.0f;
        headsBF[gid] = (__bf16)v;
    }
    if (gid < NB * NOP * (NCP - NC)) {   // 64*272*15 pad elements
        int c15 = gid % (NCP - NC);
        int bo  = gid / (NCP - NC);
        wsW[(size_t)bo * NCP + NC + c15] = (__bf16)0.0f;
    }
}

// ---------------------------------------------------------------------------
// Kernel 1: per 16-row tile of (b,c): fourier emb (angle-addition trick,
// 128 transcendentals per row instead of 2048), scores GEMM via WMMA bf16,
// softmax over 270 heads, store weights bf16 transposed as wsW[b][o][c]
// (c padded to 288; pad zeroed by kernel 0).
// ---------------------------------------------------------------------------
#define ASTRIDE 2056   // 2048 + 8 bf16 pad -> row stride 4112B == 16 mod 256 (bank spread)

__global__ __launch_bounds__(256) void sa_scores_softmax_kernel(
    const float* __restrict__ positions, const __bf16* __restrict__ headsBF,
    __bf16* __restrict__ wsW) {

    __shared__ __bf16 shA[16][ASTRIDE];   // emb tile, bf16 (A operand), ~65.8KB
    __shared__ float  shSC[16][128];      // per-row sin/cos tables (sx,cx,sy,cy)
    __shared__ float  shScores[16][NOP];  // fp32 scores for softmax

    const float TWO_PI = 6.283185307179586f;
    const int tid = threadIdx.x;
    const int rowbase = blockIdx.x * 16;

    // --- sin/cos tables: 16 rows x 2 axes x 32 freqs = 1024 tasks ---
    for (int id = tid; id < 1024; id += 256) {
        int row  = id >> 6;
        int rest = id & 63;
        int axis = rest >> 5;
        int f    = rest & 31;
        float p   = (positions[(size_t)(rowbase + row) * 2 + axis] + 0.1f) * (1.0f / 1.2f);
        float ang = TWO_PI * (float)f * p;
        shSC[row][axis * 64 + f]      = __sinf(ang);
        shSC[row][axis * 64 + 32 + f] = __cosf(ang);
    }
    __syncthreads();

    // --- emb tile: sin(ai+bj), cos(ai+bj) via angle addition ---
    for (int id = tid; id < 16 * 1024; id += 256) {
        int row = id >> 10;
        int ij  = id & 1023;
        int i = ij >> 5, j = ij & 31;
        float sx = shSC[row][i],      cx = shSC[row][32 + i];
        float sy = shSC[row][64 + j], cy = shSC[row][96 + j];
        shA[row][ij]        = (__bf16)(sx * cy + cx * sy);  // sin
        shA[row][1024 + ij] = (__bf16)(cx * cy - sx * sy);  // cos
    }
    __syncthreads();

    const int wave = tid >> 5, lane = tid & 31;
    const int m   = lane & 15;    // A: M index / B: N index
    const int grp = lane >> 4;

    // --- scores GEMM: 17 N-tiles across 8 waves, K = 2048 = 64 steps ---
    for (int nt = wave; nt < 17; nt += 8) {
        int o0 = nt * 16;
        f32x8 acc = {0.f, 0.f, 0.f, 0.f, 0.f, 0.f, 0.f, 0.f};
        const __bf16* hrow = headsBF + (size_t)(o0 + m) * KPOS;  // B lane n = m
        for (int ks = 0; ks < 64; ++ks) {
            int kb = ks * 32;
            // A frag (ISA layout): e0..7 -> k=grp*8+e, e8..15 -> k=16+grp*8+(e-8)
            bf16x8 alo = *(const bf16x8*)&shA[m][kb + grp * 8];
            bf16x8 ahi = *(const bf16x8*)&shA[m][kb + 16 + grp * 8];
            // B frag: same as A: e0..7 -> k=grp*8+e, e8..15 -> k=16+grp*8+(e-8)
            bf16x8 blo = *(const bf16x8*)(hrow + kb + grp * 8);
            bf16x8 bhi = *(const bf16x8*)(hrow + kb + 16 + grp * 8);
            acc = wmma_bf16(cat8(alo, ahi), cat8(blo, bhi), acc);
        }
        // D layout: vgpr r, lane -> (m = r + 8*grp, n = lane&15)
        #pragma unroll
        for (int r = 0; r < 8; ++r)
            shScores[r + grp * 8][o0 + m] = acc[r];
    }
    __syncthreads();

    // --- softmax over o (270), 2 rows per wave; write bf16 transposed ---
    for (int rr = 0; rr < 2; ++rr) {
        int row = wave * 2 + rr;
        float vmax = -3.4e38f;
        for (int o = lane; o < NO; o += 32) vmax = fmaxf(vmax, shScores[row][o]);
        #pragma unroll
        for (int off = 16; off > 0; off >>= 1) vmax = fmaxf(vmax, __shfl_xor(vmax, off, 32));
        float vsum = 0.f;
        for (int o = lane; o < NO; o += 32) vsum += __expf(shScores[row][o] - vmax);
        #pragma unroll
        for (int off = 16; off > 0; off >>= 1) vsum += __shfl_xor(vsum, off, 32);
        float inv = 1.0f / vsum;
        int grow = rowbase + row;
        int b = grow / NC, c = grow - b * NC;
        __bf16* wr = wsW + (size_t)b * NOP * NCP + c;
        for (int o = lane; o < NO; o += 32)
            wr[(size_t)o * NCP] = (__bf16)(__expf(shScores[row][o] - vmax) * inv);
    }
}

// ---------------------------------------------------------------------------
// Kernel 2: out[b,o,t] = sum_c W[b,c,o] * X[b,c,t].  Per block: one b, one
// 16-o tile, 256 t (8 waves x 2 WMMA t-tiles).  A = bf16 weights (padded),
// B = brain f32 converted in-flight (HBM traffic unchanged).
//
// The k-loop is fully unrolled with the bounds-checked tail step (ks=8)
// peeled: all 256 main-loop brain loads become immediate-offset b32 loads
// off two per-lane base pointers (max offset 255*12000B < 2^23), so the
// backend can clause them and drop per-iteration address/predicate VALU.
// ---------------------------------------------------------------------------
__global__ __launch_bounds__(256) void sa_out_gemm_kernel(
    const float* __restrict__ brain, const __bf16* __restrict__ wsW,
    float* __restrict__ out) {

    const int wave = threadIdx.x >> 5, lane = threadIdx.x & 31;
    const int m   = lane & 15;
    const int grp = lane >> 4;
    const int b  = blockIdx.z;
    const int o0 = blockIdx.y * 16;
    const int t0 = blockIdx.x * 256 + wave * 32;
    const int ta = t0 + m, tb = t0 + 16 + m;     // B/D lane n = lane&15
    const bool oka = ta < NT, okb = tb < NT;

    const __bf16* Wrow = wsW + ((size_t)b * NOP + (o0 + m)) * NCP;  // A lane m
    const float*  X    = brain + (size_t)b * NC * NT;
    // per-lane base pointers for B loads: row c = grp*16 + e (+ kb), col ta/tb
    const float* Xa = X + (size_t)(grp * 16) * NT + ta;
    const float* Xb = X + (size_t)(grp * 16) * NT + tb;

    f32x8 acc0 = {0.f, 0.f, 0.f, 0.f, 0.f, 0.f, 0.f, 0.f};
    f32x8 acc1 = {0.f, 0.f, 0.f, 0.f, 0.f, 0.f, 0.f, 0.f};

    // --- main K steps: ks = 0..7, c = kb + grp*16 + e <= 255 < 273, no check
    #pragma unroll
    for (int ks = 0; ks < 8; ++ks) {
        const int kb = ks * 32;
        bf16x8 alo = *(const bf16x8*)(Wrow + kb + grp * 16);
        bf16x8 ahi = *(const bf16x8*)(Wrow + kb + grp * 16 + 8);
        bf16x16 a = cat8(alo, ahi);

        bf16x16 b0, b1;
        #pragma unroll
        for (int e = 0; e < 16; ++e) {
            const size_t off = (size_t)(kb + e) * NT;  // constant per unrolled iter
            float x0 = oka ? Xa[off] : 0.f;
            float x1 = okb ? Xb[off] : 0.f;
            b0[e] = (__bf16)x0;
            b1[e] = (__bf16)x1;
        }
        acc0 = wmma_bf16(a, b0, acc0);
        acc1 = wmma_bf16(a, b1, acc1);
    }

    // --- peeled tail step: ks = 8, kb = 256, c in [256,287] needs c < 273
    {
        const int kb = 256;
        bf16x8 alo = *(const bf16x8*)(Wrow + kb + grp * 16);
        bf16x8 ahi = *(const bf16x8*)(Wrow + kb + grp * 16 + 8);
        bf16x16 a = cat8(alo, ahi);   // pad cols of wsW are zero-filled

        bf16x16 b0, b1;
        #pragma unroll
        for (int e = 0; e < 16; ++e) {
            const int c = kb + grp * 16 + e;
            const size_t off = (size_t)(kb + e) * NT;
            float x0 = 0.f, x1 = 0.f;
            if (c < NC) {
                if (oka) x0 = Xa[off];
                if (okb) x1 = Xb[off];
            }
            b0[e] = (__bf16)x0;
            b1[e] = (__bf16)x1;
        }
        acc0 = wmma_bf16(a, b0, acc0);
        acc1 = wmma_bf16(a, b1, acc1);
    }

    #pragma unroll
    for (int r = 0; r < 8; ++r) {               // D: m = r + 8*grp, n = lane&15
        int o = o0 + r + grp * 8;
        if (o < NO) {
            size_t base = ((size_t)b * NO + o) * (size_t)NT;
            if (oka) out[base + ta] = acc0[r];
            if (okb) out[base + tb] = acc1[r];
        }
    }
}

// ---------------------------------------------------------------------------
// Host launcher.  Workspace layout:
//   [0, 1114112)          headsBF : 272 x 2048 bf16
//   [1114112, 11141120)   wsW     : 64 x 272 x 288 bf16  (weights^T, padded)
// ---------------------------------------------------------------------------
extern "C" void kernel_launch(void* const* d_in, const int* in_sizes, int n_in,
                              void* d_out, int out_size, void* d_ws, size_t ws_size,
                              hipStream_t stream) {
    (void)in_sizes; (void)n_in; (void)out_size; (void)ws_size;

    const float* brain     = (const float*)d_in[0];   // [64,273,3000] f32
    const float* positions = (const float*)d_in[1];   // [64,273,2]    f32
    const float* heads     = (const float*)d_in[2];   // [270,2048]    f32
    float* out = (float*)d_out;                       // [64,270,3000] f32

    __bf16* headsBF = (__bf16*)d_ws;
    __bf16* wsW     = (__bf16*)((char*)d_ws + (size_t)NOP * KPOS * 2);

    // Kernel 0: conversions + pad zero-fill
    {
        int tasks = NOP * KPOS;                        // 557056 (covers pad fill too)
        sa_prep_kernel<<<(tasks + 255) / 256, 256, 0, stream>>>(heads, headsBF, wsW);
    }
    // Kernel 1: scores + softmax -> bf16 weights^T (1092 row tiles)
    sa_scores_softmax_kernel<<<NROWS / 16, 256, 0, stream>>>(positions, headsBF, wsW);

    // Kernel 2: batched GEMM out = W^T x X
    dim3 g2((NT + 255) / 256, (NOP + 15) / 16, NB);    // (12, 17, 64)
    sa_out_gemm_kernel<<<g2, 256, 0, stream>>>(brain, wsW, out);
}
